// ObserverNeuralNetwork_26319559590099
// MI455X (gfx1250) — matrix operand order repaired and checked by reading, not compiled, not run on hardware
//
#include <hip/hip_runtime.h>
#include <hip/hip_bf16.h>

// ---------------------------------------------------------------------------
// MI455X (gfx1250) implementation.
// Heavy path: [256 x K] @ [K x 2048] GEMMs in bf16 WMMA (v_wmma_f32_16x16x32_bf16),
// weights cast once per launch to bf16 (~92MB, L2-resident across the 8 beats).
// LDS staging is layout-matched to the CDNA5 WMMA VGPR layouts so both A and B
// fragments are pure ds_load_b128 (B tile is stored transposed in LDS).
// Everything else (FFT feature, stats, LayerNorm, gates, 8-qubit statevector,
// tiny projections) runs as small deterministic VALU kernels.
// ---------------------------------------------------------------------------

typedef __attribute__((ext_vector_type(16))) __bf16          v16bf;
typedef __attribute__((ext_vector_type(16))) unsigned short  v16u;
typedef __attribute__((ext_vector_type(8)))  float           v8f;

#define BATCH 256
#define HDIM  2048
#define NBEAT 8
#define BEATL 256
#define INPC  12

__device__ __forceinline__ unsigned short f2bf(float f) {
  unsigned int x = __float_as_uint(f);
  unsigned int lsb = (x >> 16) & 1u;
  x += 0x7fffu + lsb;                       // round-to-nearest-even
  return (unsigned short)(x >> 16);
}

// ------------------------------- cast f32 -> bf16 ---------------------------
__global__ void k_cast(const float* __restrict__ src, unsigned short* __restrict__ dst, int n) {
  int i = blockIdx.x * 256 + threadIdx.x;
  if (i < n) dst[i] = f2bf(src[i]);
}

// ------------------------- beat packing (time / delta) ----------------------
__global__ void k_pack_time(const float* __restrict__ x, unsigned short* __restrict__ dst, int bi) {
  int idx = blockIdx.x * 256 + threadIdx.x;                  // B * 3072
  if (idx >= BATCH * BEATL * INPC) return;
  int b = idx / (BEATL * INPC), r = idx % (BEATL * INPC);
  dst[idx] = f2bf(x[((size_t)b * 2048 + (size_t)bi * BEATL) * INPC + r]);
}

__global__ void k_pack_delta(const float* __restrict__ x, unsigned short* __restrict__ dst, int bi) {
  int idx = blockIdx.x * 256 + threadIdx.x;                  // B * 3060
  const int RL = (BEATL - 1) * INPC;
  if (idx >= BATCH * RL) return;
  int b = idx / RL, r = idx % RL, t = r / INPC, c = r % INPC;
  size_t base = ((size_t)b * 2048 + (size_t)bi * BEATL) * INPC;
  dst[idx] = f2bf(x[base + (size_t)(t + 1) * INPC + c] - x[base + (size_t)t * INPC + c]);
}

// ------------------------- freq feature: mean |rfft| ------------------------
__global__ __launch_bounds__(128) void k_freq(const float* __restrict__ x, float* __restrict__ out, int bi) {
  int b = blockIdx.x / INPC, ch = blockIdx.x % INPC, tid = threadIdx.x;
  __shared__ float s[BEATL];
  __shared__ float red[128];
  const float* base = x + ((size_t)b * 2048 + (size_t)bi * BEATL) * INPC + ch;
  s[tid]       = base[(size_t)tid * INPC];
  s[tid + 128] = base[(size_t)(tid + 128) * INPC];
  __syncthreads();
  float acc = 0.f;
  for (int k = tid; k < BEATL / 2 + 1; k += 128) {
    float w = -6.283185307179586f * (float)k / (float)BEATL;
    float re = 0.f, im = 0.f;
    for (int t = 0; t < BEATL; ++t) {
      float a = w * (float)t;
      re += s[t] * __cosf(a);
      im += s[t] * __sinf(a);
    }
    acc += sqrtf(re * re + im * im);
  }
  red[tid] = acc; __syncthreads();
  for (int o = 64; o > 0; o >>= 1) { if (tid < o) red[tid] += red[tid + o]; __syncthreads(); }
  if (tid == 0) out[b * INPC + ch] = red[0] / (float)(BEATL / 2 + 1);
}

// ------------------------- stat feature: mean / std(ddof=1) -----------------
__global__ __launch_bounds__(256) void k_stat(const float* __restrict__ x, float* __restrict__ out, int bi) {
  int b = blockIdx.x / INPC, ch = blockIdx.x % INPC, tid = threadIdx.x;
  __shared__ float red[256];
  float v = x[((size_t)b * 2048 + (size_t)bi * BEATL + tid) * INPC + ch];
  red[tid] = v; __syncthreads();
  for (int o = 128; o > 0; o >>= 1) { if (tid < o) red[tid] += red[tid + o]; __syncthreads(); }
  float mean = red[0] / (float)BEATL;
  __syncthreads();
  float d = v - mean;
  red[tid] = d * d; __syncthreads();
  for (int o = 128; o > 0; o >>= 1) { if (tid < o) red[tid] += red[tid + o]; __syncthreads(); }
  if (tid == 0) {
    out[b * 2 * INPC + ch]        = mean;
    out[b * 2 * INPC + INPC + ch] = sqrtf(red[0] / (float)(BEATL - 1));
  }
}

// ------------------------------- WMMA GEMM ----------------------------------
// C[M=256, N=2048] = act(A[M,K](bf16) @ W[K,N](bf16) + bias), f32 out.
// Block tile 64x128, 4 waves (2x2), each wave a 32x64 tile = 2x4 16x16 accs
// -> 8 v_wmma per K-step per wave between barriers.
// LDS: A row-major (frag = 2x b128), B transposed [n][k] (frag = 2x b128).
#define GTM 64
#define GTN 128
#define GTK 32

template <int ACT>
__global__ __launch_bounds__(128) void k_gemm(
    const unsigned short* __restrict__ A, const unsigned short* __restrict__ W,
    const float* __restrict__ bias, float* __restrict__ Cf, int K, int N) {
  __shared__ __align__(16) unsigned short lA[GTM][GTK + 8];   // [m][k]
  __shared__ __align__(16) unsigned short lBt[GTN][GTK + 8];  // [n][k] (transposed)
  const int tid  = threadIdx.x;
  const int bm   = blockIdx.y * GTM, bn = blockIdx.x * GTN;
  const int wid  = tid >> 5, lane = tid & 31;
  const int wm   = wid >> 1, wn = wid & 1;        // 2x2 wave grid
  const int lh   = lane >> 4, l16 = lane & 15;
  const int ar   = tid >> 1, ac = (tid & 1) * 16; // A-tile: 64 rows x 2 col-groups of 16
  const int br   = tid >> 2, bc = (tid & 3) * 32; // B-tile: 32 rows x 4 col-groups of 32
  const bool avec = (K % 8) == 0;

  v8f acc[2][4];
  for (int i = 0; i < 2; ++i)
    for (int j = 0; j < 4; ++j)
      for (int e = 0; e < 8; ++e) acc[i][j][e] = 0.f;

  for (int k0 = 0; k0 < K; k0 += GTK) {
    // ---- stage A tile (64 x 32 bf16), 16 elems per thread ----
    if (avec && (k0 + GTK) <= K) {
      const uint4* pa = reinterpret_cast<const uint4*>(A + (size_t)(bm + ar) * K + k0 + ac);
      *reinterpret_cast<uint4*>(&lA[ar][ac])     = pa[0];
      *reinterpret_cast<uint4*>(&lA[ar][ac + 8]) = pa[1];
    } else {
      for (int j = 0; j < 16; ++j) {
        int kk = k0 + ac + j;
        lA[ar][ac + j] = (kk < K) ? A[(size_t)(bm + ar) * K + kk] : (unsigned short)0;
      }
    }
    // ---- stage B tile (32 x 128 bf16) transposed, 32 elems per thread ----
    if ((k0 + br) < K) {
      unsigned short tmp[32];
      const uint4* pb = reinterpret_cast<const uint4*>(W + (size_t)(k0 + br) * N + bn + bc);
      *reinterpret_cast<uint4*>(tmp)      = pb[0];
      *reinterpret_cast<uint4*>(tmp + 8)  = pb[1];
      *reinterpret_cast<uint4*>(tmp + 16) = pb[2];
      *reinterpret_cast<uint4*>(tmp + 24) = pb[3];
      for (int j = 0; j < 32; ++j) lBt[bc + j][br] = tmp[j];
    } else {
      for (int j = 0; j < 32; ++j) lBt[bc + j][br] = 0;
    }
    if (k0 + GTK < K) {   // hint next K tile into cache (global_prefetch_b8)
      __builtin_prefetch(A + (size_t)(bm + ar) * K + k0 + GTK + ac, 0, 1);
      __builtin_prefetch(W + (size_t)(k0 + GTK + br) * N + bn + bc, 0, 1);
    }
    __syncthreads();

    // ---- fragments straight out of LDS as b128 pairs ----
    // A frag: lane (lh,l16): elements 0..7  = lA[row][lh*8 .. +7]
    //                        elements 8..15 = lA[row][lh*8+16 .. +7]
    v16bf af[2];
    for (int tm = 0; tm < 2; ++tm) {
      int row = wm * 32 + tm * 16 + l16;
      int kb = lh * 8;
      union { v16u v; uint4 q[2]; } ua;
      ua.q[0] = *reinterpret_cast<const uint4*>(&lA[row][kb]);
      ua.q[1] = *reinterpret_cast<const uint4*>(&lA[row][kb + 16]);
      af[tm] = __builtin_bit_cast(v16bf, ua.v);
    }
    // B frag: lane (lh,l16): elements 0..7 = B[K = lh*8 + j][col], 8..15 = B[K = lh*8+16 + (j-8)][col]
    v16bf bfv[4];
    for (int tn = 0; tn < 4; ++tn) {
      int col = wn * 64 + tn * 16 + l16;
      int kb = lh * 8;
      union { v16u v; uint4 q[2]; } ub;
      ub.q[0] = *reinterpret_cast<const uint4*>(&lBt[col][kb]);
      ub.q[1] = *reinterpret_cast<const uint4*>(&lBt[col][kb + 16]);
      bfv[tn] = __builtin_bit_cast(v16bf, ub.v);
    }
    for (int tm = 0; tm < 2; ++tm)
      for (int tn = 0; tn < 4; ++tn)
        acc[tm][tn] = __builtin_amdgcn_wmma_f32_16x16x32_bf16(
            false, af[tm], false, bfv[tn], (short)0, acc[tm][tn], false, false);
    __syncthreads();
  }

  // ---- epilogue: bias + activation, f32 store ----
  for (int tm = 0; tm < 2; ++tm)
    for (int tn = 0; tn < 4; ++tn) {
      int col = bn + wn * 64 + tn * 16 + l16;
      float bv = bias[col];
      for (int v = 0; v < 8; ++v) {
        int row = bm + wm * 32 + tm * 16 + lh * 8 + v;
        float o = acc[tm][tn][v] + bv;
        if (ACT == 1) o = 0.5f * o * (1.f + erff(o * 0.70710678118654752f)); // exact GELU
        Cf[(size_t)row * N + col] = o;
      }
    }
}

// ------------------------------- LayerNorm ----------------------------------
__global__ __launch_bounds__(256) void k_layernorm(
    const float* __restrict__ in, const float* __restrict__ g, const float* __restrict__ b,
    float* __restrict__ outf, unsigned short* __restrict__ outbf, float scale) {
  int row = blockIdx.x, tid = threadIdx.x;
  __shared__ float red[256];
  __shared__ float sm, sv;
  const float* p = in + (size_t)row * HDIM;
  float vals[HDIM / 256];
  float s = 0.f;
  for (int j = 0; j < HDIM / 256; ++j) { vals[j] = p[tid + j * 256] * scale; s += vals[j]; }
  red[tid] = s; __syncthreads();
  for (int o = 128; o > 0; o >>= 1) { if (tid < o) red[tid] += red[tid + o]; __syncthreads(); }
  if (tid == 0) sm = red[0] / (float)HDIM;
  __syncthreads();
  float m = sm; s = 0.f;
  for (int j = 0; j < HDIM / 256; ++j) { float d = vals[j] - m; s += d * d; }
  red[tid] = s; __syncthreads();
  for (int o = 128; o > 0; o >>= 1) { if (tid < o) red[tid] += red[tid + o]; __syncthreads(); }
  if (tid == 0) sv = rsqrtf(red[0] / (float)HDIM + 1e-5f);
  __syncthreads();
  float inv = sv;
  for (int j = 0; j < HDIM / 256; ++j) {
    int c = tid + j * 256;
    float o = (vals[j] - m) * inv * g[c] + b[c];
    if (outf)  outf[(size_t)row * HDIM + c]  = o;
    if (outbf) outbf[(size_t)row * HDIM + c] = f2bf(o);
  }
}

// -------------------------- elementwise helpers -----------------------------
__global__ void k_fuse(const float* __restrict__ a, const float* __restrict__ b,
                       const float* __restrict__ c, const float* __restrict__ d,
                       const float* __restrict__ dw, float* __restrict__ out, int n) {
  int i = blockIdx.x * 256 + threadIdx.x;
  if (i >= n) return;
  float w0 = dw[0], w1 = dw[1], w2 = dw[2], w3 = dw[3];
  float mx = fmaxf(fmaxf(w0, w1), fmaxf(w2, w3));
  float e0 = expf(w0 - mx), e1 = expf(w1 - mx), e2 = expf(w2 - mx), e3 = expf(w3 - mx);
  float inv = 1.f / (e0 + e1 + e2 + e3);
  out[i] = (e0 * a[i] + e1 * b[i] + e2 * c[i] + e3 * d[i]) * inv;
}

__global__ void k_combine(const float* __restrict__ h, const float* __restrict__ gl,
                          const float* __restrict__ c, float* __restrict__ out,
                          float alpha, int n) {
  int i = blockIdx.x * 256 + threadIdx.x;
  if (i >= n) return;
  float g = 1.f / (1.f + expf(-gl[i]));
  out[i] = h[i] + alpha * g * c[i];
}

__global__ void k_accum(float* __restrict__ acc, const float* __restrict__ h, float beta, int n) {
  int i = blockIdx.x * 256 + threadIdx.x;
  if (i < n) acc[i] = beta * acc[i] + h[i];
}

// small-K dense: out[b,n] = bias[n] + sum_k A[b,k] * W[k,n]
__global__ void k_smallk(const float* __restrict__ A, int K, const float* __restrict__ W,
                         const float* __restrict__ bias, float* __restrict__ out, int N) {
  int idx = blockIdx.x * 256 + threadIdx.x;
  if (idx >= BATCH * N) return;
  int b = idx / N, n = idx % N;
  float s = bias[n];
  for (int k = 0; k < K; ++k) s += A[b * K + k] * W[k * N + n];
  out[idx] = s;
}

// angles = tanh(h @ pinW + pinb) * pi + phase      (h:[B,2048] -> [B,8])
__global__ __launch_bounds__(256) void k_pin(const float* __restrict__ h,
                                             const float* __restrict__ pw,
                                             const float* __restrict__ pb,
                                             float phase, float* __restrict__ ang) {
  int b = blockIdx.x, tid = threadIdx.x;
  __shared__ float red[256];
  __shared__ float a8[8];
  float part[8];
  for (int q = 0; q < 8; ++q) part[q] = 0.f;
  for (int i = tid; i < HDIM; i += 256) {
    float hv = h[(size_t)b * HDIM + i];
    const float* wr = pw + (size_t)i * 8;
    for (int q = 0; q < 8; ++q) part[q] += hv * wr[q];
  }
  for (int q = 0; q < 8; ++q) {
    red[tid] = part[q]; __syncthreads();
    for (int o = 128; o > 0; o >>= 1) { if (tid < o) red[tid] += red[tid + o]; __syncthreads(); }
    if (tid == 0) a8[q] = red[0];
    __syncthreads();
  }
  if (tid < 8) ang[b * 8 + tid] = tanhf(a8[tid] + pb[tid]) * 3.14159265358979323846f + phase;
}

// ---------------------- 8-qubit statevector observer ------------------------
__device__ __forceinline__ void q_ry(float* re, float* im, int tid, int mask, float th) {
  if ((tid & mask) == 0) {
    int p = tid | mask;
    float c = __cosf(0.5f * th), s = __sinf(0.5f * th);
    float a0 = re[tid], b0 = im[tid], a1 = re[p], b1 = im[p];
    re[tid] = c * a0 - s * a1; im[tid] = c * b0 - s * b1;
    re[p]   = s * a0 + c * a1; im[p]   = s * b0 + c * b1;
  }
  __syncthreads();
}
__device__ __forceinline__ void q_rz(float* re, float* im, int tid, int mask, float th) {
  float half = (tid & mask) ? 0.5f * th : -0.5f * th;   // diag(e^{-i t/2}, e^{+i t/2})
  float c = __cosf(half), s = __sinf(half);
  float a = re[tid], b = im[tid];
  re[tid] = a * c - b * s; im[tid] = a * s + b * c;
  __syncthreads();
}
__device__ __forceinline__ void q_cnot(float* re, float* im, int tid, int cm, int tm) {
  if ((tid & cm) && !(tid & tm)) {
    int p = tid | tm;
    float a = re[tid], b = im[tid];
    re[tid] = re[p]; im[tid] = im[p];
    re[p] = a; im[p] = b;
  }
  __syncthreads();
}

__global__ __launch_bounds__(256) void k_quantum(const float* __restrict__ angles,
                                                 const float* __restrict__ wry,
                                                 const float* __restrict__ wrz,
                                                 float* __restrict__ meas) {
  __shared__ float re[256], im[256], red[256], za[8];
  int tid = threadIdx.x, b = blockIdx.x;
  re[tid] = (tid == 0) ? 1.f : 0.f; im[tid] = 0.f;
  __syncthreads();
  for (int i = 0; i < 8; ++i) q_ry(re, im, tid, 1 << (7 - i), angles[b * 8 + i]);
  for (int i = 1; i < 8; ++i) q_cnot(re, im, tid, 1 << 7, 1 << (7 - i));       // CNOT(0,i)
  for (int l = 0; l < 2; ++l)
    for (int i = 0; i < 8; ++i) {
      q_ry(re, im, tid, 1 << (7 - i), wry[l * 8 + i]);
      q_rz(re, im, tid, 1 << (7 - i), wrz[l * 8 + i]);
    }
  for (int i = 1; i < 8; ++i) q_cnot(re, im, tid, 1 << (7 - i), 1 << 7);       // CNOT(i,0)
  float p = re[tid] * re[tid] + im[tid] * im[tid];
  for (int q = 0; q < 8; ++q) {                       // deterministic tree reductions
    red[tid] = ((tid >> (7 - q)) & 1) ? -p : p;
    __syncthreads();
    for (int o = 128; o > 0; o >>= 1) { if (tid < o) red[tid] += red[tid + o]; __syncthreads(); }
    if (tid == 0) za[q] = red[0];
    __syncthreads();
  }
  if (tid < 8) meas[b * 8 + tid] = za[tid];
}

// ------------------------------ head (H -> 5) -------------------------------
__global__ __launch_bounds__(256) void k_head(const float* __restrict__ ln,
                                              const float* __restrict__ W,
                                              const float* __restrict__ bias,
                                              float* __restrict__ out) {
  int b = threadIdx.x;                                // single block of 256
  float acc[5];
  for (int o = 0; o < 5; ++o) acc[o] = bias[o];
  for (int h = 0; h < HDIM; ++h) {
    float v = ln[(size_t)b * HDIM + h];
    const float* wr = W + (size_t)h * 5;
    for (int o = 0; o < 5; ++o) acc[o] += v * wr[o];
  }
  for (int o = 0; o < 5; ++o) out[b * 5 + o] = acc[o];
}

// ---------------------------------------------------------------------------
// Host orchestration. Input indices follow alphabetical pytree flatten of
// setup_inputs(): params(fusion{delta,dw,freq,norm,stat,time}, head, inner,
// ion, norm_out, obs) then x.  lin keys: W,b ; ln keys: b,g.
// ---------------------------------------------------------------------------
extern "C" void kernel_launch(void* const* d_in, const int* in_sizes, int n_in,
                              void* d_out, int out_size, void* d_ws, size_t ws_size,
                              hipStream_t stream) {
  (void)in_sizes; (void)n_in; (void)out_size; (void)ws_size;
  auto F = [&](int i) { return (const float*)d_in[i]; };

  // ---- parameter index map (alphabetical flatten) ----
  // fusion.delta: l1.W=0 l1.b=1 l2.W=2 l2.b=3 ln.b=4 ln.g=5 ; dw=6
  // fusion.freq : l1.W=7 l1.b=8 l2.W=9 l2.b=10 ln.b=11 ln.g=12
  // fusion.norm : b=13 g=14
  // fusion.stat : l1.W=15 l1.b=16 l2.W=17 l2.b=18 ln.b=19 ln.g=20
  // fusion.time : l1.W=21 l1.b=22 l2.W=23 l2.b=24 ln.b=25 ln.g=26
  // head: W=27 b=28
  // inner: dec_l1.W=29 b=30 dec_l2.W=31 b=32 dec_ln.b=33 g=34 gate.W=35 b=36
  //        n1.b=37 g=38 n2.b=39 g=40
  // ion: gate.W=41 b=42 norm.b=43 g=44 proj.W=45 b=46
  // norm_out: b=47 g=48
  // obs: norm.b=49 g=50 pin.W=51 b=52 pout.W=53 b=54 wry=55 wrz=56 ; x=57
  const float* x = F(57);

  size_t off = 0;
  auto wsAlloc = [&](size_t bytes) -> void* {
    void* p = (char*)d_ws + off;
    off += bytes; off = (off + 255) & ~(size_t)255;
    return p;
  };
  auto castW = [&](int idx, size_t n) -> unsigned short* {
    unsigned short* p = (unsigned short*)wsAlloc(n * 2);
    k_cast<<<dim3((unsigned)((n + 255) / 256)), dim3(256), 0, stream>>>(F(idx), p, (int)n);
    return p;
  };

  // ---- bf16 weight copies (L2-resident across all 8 beats) ----
  unsigned short* wTimeL1  = castW(21, (size_t)3072 * HDIM);
  unsigned short* wTimeL2  = castW(23, (size_t)HDIM * HDIM);
  unsigned short* wDeltaL1 = castW(0,  (size_t)3060 * HDIM);
  unsigned short* wDeltaL2 = castW(2,  (size_t)HDIM * HDIM);
  unsigned short* wFreqL1  = castW(7,  (size_t)12 * HDIM);
  unsigned short* wFreqL2  = castW(9,  (size_t)HDIM * HDIM);
  unsigned short* wStatL1  = castW(15, (size_t)24 * HDIM);
  unsigned short* wStatL2  = castW(17, (size_t)HDIM * HDIM);
  unsigned short* wDecL1   = castW(29, (size_t)HDIM * HDIM);
  unsigned short* wDecL2   = castW(31, (size_t)HDIM * HDIM);
  unsigned short* wInGate  = castW(35, (size_t)HDIM * HDIM);
  unsigned short* wIonGate = castW(41, (size_t)HDIM * HDIM);

  // ---- activations ----
  const size_t BH = (size_t)BATCH * HDIM;
  unsigned short* timeIn  = (unsigned short*)wsAlloc((size_t)BATCH * 3072 * 2);
  unsigned short* deltaIn = (unsigned short*)wsAlloc((size_t)BATCH * 3060 * 2);
  float* freqF  = (float*)wsAlloc((size_t)BATCH * 12 * 4);
  unsigned short* freqBf = (unsigned short*)wsAlloc((size_t)BATCH * 12 * 2);
  float* statF  = (float*)wsAlloc((size_t)BATCH * 24 * 4);
  unsigned short* statBf = (unsigned short*)wsAlloc((size_t)BATCH * 24 * 2);
  float* encTmp = (float*)wsAlloc(BH * 4);
  unsigned short* encLnBf = (unsigned short*)wsAlloc(BH * 2);
  float* fT = (float*)wsAlloc(BH * 4);
  float* fFq = (float*)wsAlloc(BH * 4);
  float* fS = (float*)wsAlloc(BH * 4);
  float* fD = (float*)wsAlloc(BH * 4);
  float* preF = (float*)wsAlloc(BH * 4);
  float* hF = (float*)wsAlloc(BH * 4);
  unsigned short* hBf = (unsigned short*)wsAlloc(BH * 2);
  float* gateLin = (float*)wsAlloc(BH * 4);
  float* ctxPre = (float*)wsAlloc(BH * 4);
  float* ctxF = (float*)wsAlloc(BH * 4);
  unsigned short* h1Bf = (unsigned short*)wsAlloc(BH * 2);
  float* d1 = (float*)wsAlloc(BH * 4);
  unsigned short* dlnBf = (unsigned short*)wsAlloc(BH * 2);
  float* d2 = (float*)wsAlloc(BH * 4);
  float* ssum = (float*)wsAlloc(BH * 4);
  float* lnout = (float*)wsAlloc(BH * 4);
  float* logits = (float*)wsAlloc((size_t)BATCH * 5 * 4);
  float* angles = (float*)wsAlloc((size_t)BATCH * 8 * 4);
  float* meas   = (float*)wsAlloc((size_t)BATCH * 8 * 4);

  const dim3 gGemm(HDIM / GTN, BATCH / GTM);  // (16, 4)
  const unsigned EW = (unsigned)((BH + 255) / 256);
  const float PH[3] = {0.f, 2.0943951023931953f, 4.1887902047863905f};

  auto gemm = [&](const unsigned short* Abf, int K, const unsigned short* Wbf,
                  const float* bias, int act, float* outF) {
    if (act == 1) k_gemm<1><<<gGemm, 128, 0, stream>>>(Abf, Wbf, bias, outF, K, HDIM);
    else          k_gemm<0><<<gGemm, 128, 0, stream>>>(Abf, Wbf, bias, outF, K, HDIM);
  };
  auto lnorm = [&](const float* in, int gi, int bi_, float* outF, unsigned short* outBf, float scale) {
    k_layernorm<<<BATCH, 256, 0, stream>>>(in, F(gi), F(bi_), outF, outBf, scale);
  };
  auto runEnc = [&](const unsigned short* Abf, int K, const unsigned short* W1, int b1,
                    int lnB, int lnG, const unsigned short* W2, int b2, float* outF) {
    gemm(Abf, K, W1, F(b1), 1, encTmp);                    // l1 + GELU
    lnorm(encTmp, lnG, lnB, nullptr, encLnBf, 1.f);        // LN -> bf16
    gemm(encLnBf, HDIM, W2, F(b2), 1, outF);               // l2 + GELU
  };

  for (int bi = 0; bi < NBEAT; ++bi) {
    // ---- fusion features ----
    k_pack_time<<<(BATCH * 3072 + 255) / 256, 256, 0, stream>>>(x, timeIn, bi);
    k_pack_delta<<<(BATCH * 3060 + 255) / 256, 256, 0, stream>>>(x, deltaIn, bi);
    k_freq<<<BATCH * INPC, 128, 0, stream>>>(x, freqF, bi);
    k_cast<<<(BATCH * 12 + 255) / 256, 256, 0, stream>>>(freqF, freqBf, BATCH * 12);
    k_stat<<<BATCH * INPC, 256, 0, stream>>>(x, statF, bi);
    k_cast<<<(BATCH * 24 + 255) / 256, 256, 0, stream>>>(statF, statBf, BATCH * 24);

    runEnc(timeIn, 3072, wTimeL1, 22, 25, 26, wTimeL2, 24, fT);
    runEnc(freqBf, 12,   wFreqL1, 8,  11, 12, wFreqL2, 10, fFq);
    runEnc(statBf, 24,   wStatL1, 16, 19, 20, wStatL2, 18, fS);
    runEnc(deltaIn, 3060, wDeltaL1, 1, 4, 5, wDeltaL2, 3, fD);

    k_fuse<<<EW, 256, 0, stream>>>(fT, fFq, fS, fD, F(6), preF, (int)BH);
    lnorm(preF, 14, 13, hF, hBf, 1.f);                     // fusion.norm

    // ---- ion gate (beats 1..7) ----
    if (bi > 0) {
      gemm(hBf, HDIM, wIonGate, F(42), 0, gateLin);
      k_smallk<<<(unsigned)((BH + 255) / 256), 256, 0, stream>>>(logits, 5, F(45), F(46), ctxPre, HDIM);
      k_combine<<<EW, 256, 0, stream>>>(hF, gateLin, ctxPre, preF, 1.f, (int)BH);
      lnorm(preF, 44, 43, hF, hBf, 1.f);                   // ion.norm
    }

    // ---- observer + decode, NH=3 ----
    for (int k = 0; k < 3; ++k) {
      k_pin<<<BATCH, 256, 0, stream>>>(hF, F(51), F(52), PH[k], angles);
      k_quantum<<<BATCH, 256, 0, stream>>>(angles, F(55), F(56), meas);
      k_smallk<<<(unsigned)((BH + 255) / 256), 256, 0, stream>>>(meas, 8, F(53), F(54), ctxPre, HDIM);
      lnorm(ctxPre, 50, 49, ctxF, nullptr, 1.f);           // obs.norm
      gemm(hBf, HDIM, wInGate, F(36), 0, gateLin);
      k_combine<<<EW, 256, 0, stream>>>(hF, gateLin, ctxF, preF, (float)(k + 1) / 3.f, (int)BH);
      lnorm(preF, 38, 37, nullptr, h1Bf, 1.f);             // n1
      gemm(h1Bf, HDIM, wDecL1, F(30), 1, d1);              // dec_l1 + GELU
      lnorm(d1, 34, 33, nullptr, dlnBf, 1.f);              // dec_ln
      gemm(dlnBf, HDIM, wDecL2, F(32), 0, d2);             // dec_l2
      lnorm(d2, 40, 39, hF, hBf, 1.f);                     // n2 -> new h
    }

    // ---- state accumulation + per-beat logits ----
    k_accum<<<EW, 256, 0, stream>>>(ssum, hF, (bi == 0) ? 0.f : 1.f, (int)BH);
    lnorm(hF, 48, 47, lnout, nullptr, 1.f);                // norm_out
    k_head<<<1, 256, 0, stream>>>(lnout, F(27), F(28), logits);
  }

  // ---- final: LN(mean of states) -> head -> d_out ----
  lnorm(ssum, 48, 47, lnout, nullptr, 1.f / (float)NBEAT);
  k_head<<<1, 256, 0, stream>>>(lnout, F(27), F(28), (float*)d_out);
}
